// SemanticComposite_42949672961824
// MI455X (gfx1250) — matrix operand order repaired and checked by reading, not compiled, not run on hardware
//
#include <hip/hip_runtime.h>
#include <stdint.h>

#define BDIM 8
#define SDIM 2048
#define FDIM 256
#define GDIM 512
#define NW   4
#define ROWS_PER_WG (16 * NW)

typedef __attribute__((ext_vector_type(16))) __bf16 bh16;
typedef __attribute__((ext_vector_type(8)))  __bf16 bh8;
typedef __attribute__((ext_vector_type(8)))  float  v8f;
typedef __attribute__((ext_vector_type(4)))  unsigned int u32x4;
typedef __attribute__((ext_vector_type(8)))  int i32x8;
typedef __attribute__((ext_vector_type(4)))  int i32x4;

static __device__ __forceinline__ __bf16 f2bf(float f) {
  unsigned u = __float_as_uint(f);
  unsigned r = u + 0x7FFFu + ((u >> 16) & 1u);
  unsigned short h = (unsigned short)(r >> 16);
  return __builtin_bit_cast(__bf16, h);
}

static __device__ __forceinline__ bh8 ld8(const __bf16* p) {
  return *reinterpret_cast<const bh8*>(p);
}
static __device__ __forceinline__ bh16 join8(bh8 lo, bh8 hi) {
  return __builtin_shufflevector(lo, hi, 0,1,2,3,4,5,6,7,8,9,10,11,12,13,14,15);
}

// A fragment (16x32 bf16): lanes 0-15 row=lr K={k0+0..7, k0+16..23},
// lanes 16-31 row=lr K={k0+8..15, k0+24..31}
static __device__ __forceinline__ bh16 loadA(const __bf16* base, int stride, int row0, int k0) {
  int lane = threadIdx.x & 31;
  int lr = lane & 15;
  int hi = lane >> 4;
  const __bf16* p = base + (size_t)(row0 + lr) * stride + k0 + hi * 8;
  return join8(ld8(p), ld8(p + 16));
}
// B fragment (32x16): lane col n = n0+lr, K = {k0+8h+0..7, k0+16+8h+0..7} (h = lane>>4),
// where element B[k][n] lives at base[(n0+lr)*stride + k]  ("n"-major storage)
static __device__ __forceinline__ bh16 loadB(const __bf16* base, int stride, int n0, int k0) {
  int lane = threadIdx.x & 31;
  int lr = lane & 15;
  int hi = lane >> 4;
  const __bf16* p = base + (size_t)(n0 + lr) * stride + k0 + hi * 8;
  return join8(ld8(p), ld8(p + 16));
}

static __device__ __forceinline__ v8f wmma_bf16(bh16 a, bh16 b, v8f c) {
  return __builtin_amdgcn_wmma_f32_16x16x32_bf16(false, a, false, b, (short)0, c, false, false);
}
static __device__ __forceinline__ v8f vzero8() {
  v8f v;
#pragma unroll
  for (int i = 0; i < 8; ++i) v[i] = 0.0f;
  return v;
}

// TDM: async DMA of one 32x256 bf16 tile (row-major) from global into LDS
static __device__ __forceinline__ void tdm_load_tile(const __bf16* gsrc, __bf16* ldst) {
  unsigned long long ga = (unsigned long long)(const void*)gsrc;
  unsigned ldsOff = (unsigned)(unsigned long long)(const void*)ldst;
  u32x4 g0;
  g0[0] = 1u;                     // count=1, user descriptor
  g0[1] = ldsOff;                 // lds_addr
  g0[2] = (unsigned)ga;           // global_addr[31:0]
  g0[3] = ((unsigned)(ga >> 32) & 0x01FFFFFFu) | (2u << 30);  // addr[56:32] | type=2
  i32x8 g1;
  g1[0] = (int)(1u << 16);        // data_size = 2 bytes
  g1[1] = (int)(256u << 16);      // tensor_dim0 = 256
  g1[2] = (int)(32u << 16);       // tensor_dim1 = 32
  g1[3] = (int)(256u << 16);      // tile_dim0 = 256
  g1[4] = 32;                     // tile_dim1 = 32, tile_dim2 = 0
  g1[5] = 256;                    // tensor_dim0_stride = 256
  g1[6] = 0;
  g1[7] = 0;
  i32x4 gz;
#pragma unroll
  for (int q = 0; q < 4; ++q) gz[q] = 0;
#if defined(__clang_major__) && __clang_major__ >= 23
  i32x8 gz8;
#pragma unroll
  for (int q = 0; q < 8; ++q) gz8[q] = 0;
  __builtin_amdgcn_tensor_load_to_lds(g0, g1, gz, gz, gz8, 0);
#else
  __builtin_amdgcn_tensor_load_to_lds(g0, g1, gz, gz, 0);
#endif
}

// ---------------- Kernel 1: per-row prep -------------------------------------
__global__ void __launch_bounds__(256) prep_rows(const float* __restrict__ x,
                                                 const float* __restrict__ attw,
                                                 __bf16* __restrict__ xbf,
                                                 __bf16* __restrict__ xwc,
                                                 float* __restrict__ av,
                                                 float* __restrict__ bv) {
  int wave = threadIdx.x >> 5;
  int lane = threadIdx.x & 31;
  int row = blockIdx.x * 8 + wave;
  const float* xr = x + (size_t)row * FDIM + lane * 8;
  float v[8];
  float sa = 0.f, sb = 0.f;
#pragma unroll
  for (int i = 0; i < 8; ++i) {
    v[i] = xr[i];
    sa += v[i] * attw[lane * 8 + i];
    sb += v[i] * attw[FDIM + lane * 8 + i];
  }
#pragma unroll
  for (int m = 16; m >= 1; m >>= 1) {
    sa += __shfl_xor(sa, m, 32);
    sb += __shfl_xor(sb, m, 32);
  }
  if (lane == 0) { av[row] = sa; bv[row] = sb; }
  __align__(16) __bf16 xb[8];
  __align__(16) __bf16 xw[8];
#pragma unroll
  for (int i = 0; i < 8; ++i) {
    xb[i] = f2bf(v[i]);
    xw[i] = f2bf(v[i] * attw[2 * FDIM + lane * 8 + i]);
  }
  *(bh8*)(xbf + (size_t)row * FDIM + lane * 8) = *(bh8*)xb;
  *(bh8*)(xwc + (size_t)row * FDIM + lane * 8) = *(bh8*)xw;
}

// ---------------- Kernel 2: weight conversion --------------------------------
__global__ void __launch_bounds__(256) prep_w(const float* __restrict__ zw,
                                              const float* __restrict__ rw,
                                              const float* __restrict__ fw,
                                              __bf16* __restrict__ wz,
                                              __bf16* __restrict__ wr,
                                              __bf16* __restrict__ wf) {
  int idx = blockIdx.x * blockDim.x + threadIdx.x;
  if (idx < FDIM * GDIM) {
    wz[idx] = f2bf(zw[idx]);
    wr[idx] = f2bf(rw[idx]);
    wf[idx] = f2bf(fw[idx]);
  }
}

// ---------------- Kernel 3: 32x32 tiled transpose xbf -> xT[b][f][s] ---------
__global__ void __launch_bounds__(256) transpose_x(const __bf16* __restrict__ xbf,
                                                   __bf16* __restrict__ xT) {
  __shared__ __bf16 tile[32][33];
  const int tilesPerBatch = (SDIM / 32) * (FDIM / 32);
  int bb = blockIdx.x / tilesPerBatch;
  int rem = blockIdx.x % tilesPerBatch;
  int j0 = (rem / (FDIM / 32)) * 32;
  int f0 = (rem % (FDIM / 32)) * 32;
  const __bf16* src = xbf + (size_t)bb * SDIM * FDIM;
  __bf16* dst = xT + (size_t)bb * SDIM * FDIM;
  for (int idx = threadIdx.x; idx < 1024; idx += 256) {
    int r = idx >> 5, c = idx & 31;
    tile[r][c] = src[(size_t)(j0 + r) * FDIM + f0 + c];
  }
  __syncthreads();
  for (int idx = threadIdx.x; idx < 1024; idx += 256) {
    int fr = idx >> 5, jc = idx & 31;
    dst[(size_t)(f0 + fr) * SDIM + j0 + jc] = tile[jc][fr];
  }
}

// ---------------- Kernel 4: fused flash-attention + gated MLP ---------------
__global__ void __launch_bounds__(128, 1) fused_attn_mlp(
    const float* __restrict__ x, const __bf16* __restrict__ xbf,
    const __bf16* __restrict__ xwc, const __bf16* __restrict__ xT,
    const float* __restrict__ av, const float* __restrict__ bv,
    const __bf16* __restrict__ wz, const __bf16* __restrict__ wr,
    const __bf16* __restrict__ wf, const float* __restrict__ zb,
    const float* __restrict__ rb, const float* __restrict__ fb,
    float* __restrict__ out) {
  // 36KB: double-buffered x tile (2 x 16KB) + P reshape (4KB); attn overlaps tiles
  __shared__ __align__(16) __bf16 smem[2 * 8192 + NW * 16 * 32];
  __bf16* xjbuf = smem;           // [2][32][256]
  __bf16* attnL = smem;           // [NW][16][256] phase-2 (overlaps tile buffers)
  __bf16* pL    = smem + 16384;   // [NW][16][32]

  const int wid  = threadIdx.x >> 5;
  const int lane = threadIdx.x & 31;
  const int lr   = lane & 15;
  const int lhi  = lane >> 4;
  const int bb   = blockIdx.x / (SDIM / ROWS_PER_WG);
  const int tb   = blockIdx.x % (SDIM / ROWS_PER_WG);
  const int row0 = tb * ROWS_PER_WG + wid * 16;
  const size_t batchOff = (size_t)bb * SDIM * FDIM;
  const __bf16* xbfB = xbf + batchOff;
  const __bf16* xwcB = xwc + batchOff;
  const __bf16* xTB  = xT + batchOff;   // [256][2048] feature-major

  float aRow[8], mRow[8], lRow[8];
#pragma unroll
  for (int e = 0; e < 8; ++e) {
    aRow[e] = av[bb * SDIM + row0 + e + lhi * 8];
    mRow[e] = -3.0e38f;
    lRow[e] = 0.0f;
  }

  // preload A fragments of (x * wc) for this wave's 16 rows (K = 256)
  bh16 aFrag[8];
#pragma unroll
  for (int kk = 0; kk < 8; ++kk) aFrag[kk] = loadA(xwcB, FDIM, row0, kk * 32);

  v8f accO[16];
#pragma unroll
  for (int t = 0; t < 16; ++t) accO[t] = vzero8();

  // prologue: kick DMA for the first tile
  if (wid == 0) tdm_load_tile(xbfB, xjbuf);

#pragma unroll 1
  for (int j0 = 0; j0 < SDIM; j0 += 32) {
    const int buf = (j0 >> 5) & 1;
    if (wid == 0) __builtin_amdgcn_s_wait_tensorcnt(0);  // current tile landed
    __syncthreads();                                     // publish; prev compute done
    if (wid == 0 && (j0 + 32) < SDIM)                    // overlap next DMA w/ compute
      tdm_load_tile(xbfB + (size_t)(j0 + 32) * FDIM, xjbuf + (buf ^ 1) * 8192);
    const __bf16* xjc = xjbuf + buf * 8192;

    // ---- scores: two 16x16 tiles, K=256, B from LDS tile ----
    v8f s0 = vzero8(), s1 = vzero8();
#pragma unroll
    for (int kk = 0; kk < 8; ++kk) {
      bh16 b0 = loadB(xjc, FDIM, 0, kk * 32);
      bh16 b1 = loadB(xjc, FDIM, 16, kk * 32);
      s0 = wmma_bf16(aFrag[kk], b0, s0);
      s1 = wmma_bf16(aFrag[kk], b1, s1);
    }

    // ---- online softmax over the 32 new columns ----
    float bj0 = bv[bb * SDIM + j0 + lr];
    float bj1 = bv[bb * SDIM + j0 + 16 + lr];
    float p0v[8], p1v[8], scl[8];
#pragma unroll
    for (int e = 0; e < 8; ++e) {
      float t0 = s0[e] + aRow[e] + bj0;
      float t1 = s1[e] + aRow[e] + bj1;
      float mx = fmaxf(t0, t1);
#pragma unroll
      for (int m = 8; m >= 1; m >>= 1) mx = fmaxf(mx, __shfl_xor(mx, m, 16));
      float mnew = fmaxf(mRow[e], mx);
      float sc = __expf(mRow[e] - mnew);
      float p0 = __expf(t0 - mnew);
      float p1 = __expf(t1 - mnew);
      float rs = p0 + p1;
#pragma unroll
      for (int m = 8; m >= 1; m >>= 1) rs += __shfl_xor(rs, m, 16);
      lRow[e] = lRow[e] * sc + rs;
      mRow[e] = mnew;
      p0v[e] = p0; p1v[e] = p1; scl[e] = sc;
    }
#pragma unroll
    for (int t = 0; t < 16; ++t)
#pragma unroll
      for (int e = 0; e < 8; ++e) accO[t][e] *= scl[e];

    // reshape P (C layout) -> A-fragment layout via wave-private LDS
    __bf16* pw = pL + wid * 512;
#pragma unroll
    for (int e = 0; e < 8; ++e) {
      int r = e + lhi * 8;
      pw[r * 32 + lr]      = f2bf(p0v[e]);
      pw[r * 32 + 16 + lr] = f2bf(p1v[e]);
    }
    asm volatile("s_wait_dscnt 0" ::: "memory");
    bh16 pA = loadA(pw, 32, 0, 0);

    // accO += P(16x32) @ Xj(32x256); B fragments contiguous from feature-major xT
#pragma unroll
    for (int t = 0; t < 16; ++t) {
      bh16 bF = loadB(xTB, SDIM, t * 16, j0);
      accO[t] = wmma_bf16(pA, bF, accO[t]);
    }
  }

  // normalize attention; park bf16 attn tile in LDS (overlaps dead tile buffers)
  float inv[8];
#pragma unroll
  for (int e = 0; e < 8; ++e) inv[e] = 1.0f / lRow[e];
  __syncthreads();  // all waves done with xjbuf
  __bf16* aw = attnL + wid * (16 * FDIM);
#pragma unroll
  for (int t = 0; t < 16; ++t)
#pragma unroll
    for (int e = 0; e < 8; ++e) {
      int r = e + lhi * 8;
      aw[r * FDIM + t * 16 + lr] = f2bf(accO[t][e] * inv[e]);
    }
  asm volatile("s_wait_dscnt 0" ::: "memory");

  // ---- gated MLP: hoist all 16 cat A-fragments (x | attn) into registers ----
  bh16 catA[16];
#pragma unroll
  for (int gs = 0; gs < 8; ++gs) catA[gs] = loadA(xbfB, FDIM, row0, gs * 32);
#pragma unroll
  for (int gs = 8; gs < 16; ++gs) catA[gs] = loadA(aw, FDIM, 0, (gs - 8) * 32);

#pragma unroll 1
  for (int ft = 0; ft < 16; ++ft) {
    if (ft + 1 < 16) {  // prefetch next weight strip (global_prefetch_b8)
      __builtin_prefetch(wz + (size_t)((ft + 1) * 16 + lr) * GDIM + lhi * 256, 0, 1);
      __builtin_prefetch(wr + (size_t)((ft + 1) * 16 + lr) * GDIM + lhi * 256, 0, 1);
      __builtin_prefetch(wf + (size_t)((ft + 1) * 16 + lr) * GDIM + lhi * 256, 0, 1);
    }
    v8f az = vzero8(), ar = vzero8(), af = vzero8();
#pragma unroll
    for (int gs = 0; gs < 16; ++gs) {
      int g0 = gs * 32;
      bh16 bz = loadB(wz, GDIM, ft * 16, g0);
      bh16 br = loadB(wr, GDIM, ft * 16, g0);
      bh16 bf_ = loadB(wf, GDIM, ft * 16, g0);
      az = wmma_bf16(catA[gs], bz, az);
      ar = wmma_bf16(catA[gs], br, ar);
      af = wmma_bf16(catA[gs], bf_, af);
    }
    int col = ft * 16 + lr;
    float zbc = zb[col], rbc = rb[col], fbc = fb[col];
#pragma unroll
    for (int e = 0; e < 8; ++e) {
      int r = row0 + e + lhi * 8;
      float xv = x[batchOff + (size_t)r * FDIM + col];
      float zv = tanhf(az[e] + zbc);
      float rv = 1.0f / (1.0f + __expf(-(ar[e] + rbc)));
      float fv = 1.0f / (1.0f + __expf(-(af[e] + fbc)));
      out[batchOff + (size_t)r * FDIM + col] = rv * xv + fv * zv;
    }
  }
}

// ---------------- launcher ---------------------------------------------------
extern "C" void kernel_launch(void* const* d_in, const int* in_sizes, int n_in,
                              void* d_out, int out_size, void* d_ws, size_t ws_size,
                              hipStream_t stream) {
  (void)in_sizes; (void)n_in; (void)out_size; (void)ws_size;
  const float* x    = (const float*)d_in[0];
  const float* attw = (const float*)d_in[1];
  const float* zw   = (const float*)d_in[2];
  const float* zb   = (const float*)d_in[3];
  const float* rw   = (const float*)d_in[4];
  const float* rb   = (const float*)d_in[5];
  const float* fw   = (const float*)d_in[6];
  const float* fb   = (const float*)d_in[7];

  char* ws = (char*)d_ws;
  __bf16* xbf = (__bf16*)(ws);                     // 8 MiB
  __bf16* xwc = (__bf16*)(ws + 8388608);           // 8 MiB
  __bf16* xT  = (__bf16*)(ws + 16777216);          // 8 MiB (feature-major)
  __bf16* wzb = (__bf16*)(ws + 25165824);          // 256 KiB
  __bf16* wrb = (__bf16*)(ws + 25165824 + 262144);
  __bf16* wfb = (__bf16*)(ws + 25165824 + 2 * 262144);
  float*  av  = (float*)(ws + 25165824 + 3 * 262144);
  float*  bvv = (float*)(ws + 25165824 + 3 * 262144 + 65536);

  prep_rows<<<(BDIM * SDIM) / 8, 256, 0, stream>>>(x, attw, xbf, xwc, av, bvv);
  prep_w<<<(FDIM * GDIM + 255) / 256, 256, 0, stream>>>(zw, rw, fw, wzb, wrb, wfb);
  transpose_x<<<BDIM * (SDIM / 32) * (FDIM / 32), 256, 0, stream>>>(xbf, xT);
  fused_attn_mlp<<<BDIM * (SDIM / ROWS_PER_WG), 128, 0, stream>>>(
      x, xbf, xwc, xT, av, bvv, wzb, wrb, wfb, zb, rb, fb, (float*)d_out);
}
